// STU_39324720562252
// MI455X (gfx1250) — matrix operand order repaired and checked by reading, not compiled, not run on hardware
//
#include <hip/hip_runtime.h>
#include <stdint.h>

#define L_IN   2048
#define D_CH   32
#define K_EIG  32
#define B_BAT  32

typedef __attribute__((ext_vector_type(16))) __bf16 v16bf;
typedef __attribute__((ext_vector_type(8)))  __bf16 v8bf;
typedef __attribute__((ext_vector_type(8)))  float  v8f;

__device__ __forceinline__ unsigned short f2bf(float f) {
    union { float f; unsigned int u; } v; v.f = f;
    unsigned int u = v.u;
    u += 0x7fffu + ((u >> 16) & 1u);          // round-to-nearest-even
    return (unsigned short)(u >> 16);
}

__device__ __forceinline__ v16bf load_v16(const unsigned short* p) {
    return *(const v16bf*)p;                   // 32B aligned -> 2x global_load_b128
}

// Gather the A-fragment pattern from row-major storage: two contiguous
// 8-element (16B) runs at p and p+16 elements (K = half*8+e | 16+half*8+e-8).
__device__ __forceinline__ v16bf load_2x8(const unsigned short* p) {
    v8bf lo = *(const v8bf*)p;
    v8bf hi = *(const v8bf*)(p + 16);
    return __builtin_shufflevector(lo, hi, 0,1,2,3,4,5,6,7,8,9,10,11,12,13,14,15);
}

__device__ __forceinline__ float bcast_lane(float v, int i) {
    union { float f; int u; } a, r;
    a.f = v;
    r.u = __builtin_amdgcn_readlane(a.u, i);   // uniform -> SGPR broadcast
    return r.f;
}

// ---------------------------------------------------------------------------
// Prep 1: bf16 copies of inputs. Ut[b][d][s] (B-fragment friendly, s-contig)
//         and Urow[b][t][i] (A-fragment friendly for the AR term).
// ---------------------------------------------------------------------------
__global__ __launch_bounds__(256)
void k_prep_u(const float* __restrict__ u, unsigned short* __restrict__ Ut,
              unsigned short* __restrict__ Urow) {
    int id = blockIdx.x * 256 + threadIdx.x;       // (b*2048+s)*32 + d
    unsigned short bf = f2bf(u[id]);
    Urow[id] = bf;
    int d  = id & 31;
    int bs = id >> 5;
    int s  = bs & (L_IN - 1);
    int b  = bs >> 11;
    Ut[((size_t)(b * D_CH + d) << 11) + s] = bf;
}

// ---------------------------------------------------------------------------
// Prep 2: bake Toeplitz A-operands directly in the WMMA A-fragment layout.
// Afrag[didx][k][lane][e] = v[16*didx + m - K(e,lane), k]  (0 if lag<0)
// didx = (t0 - s0)/16 in 0..127. 4 MB, L2-resident, reused by every batch.
// ---------------------------------------------------------------------------
__global__ __launch_bounds__(256)
void k_prep_afrag(const float* __restrict__ v, unsigned short* __restrict__ Afrag) {
    int lane = threadIdx.x & 31;
    int k    = blockIdx.y * 8 + (threadIdx.x >> 5);
    int didx = blockIdx.x;                         // delta = 16*didx
    int hl   = lane >> 4, m = lane & 15;
    unsigned short* dst = Afrag + ((((size_t)didx * K_EIG + k) * 32 + lane) << 4);
    #pragma unroll
    for (int e = 0; e < 16; ++e) {
        int K   = hl * 16 + e;
        int lag = didx * 16 + m - K;
        unsigned short val = 0;
        if (lag >= 0 && lag < L_IN) val = f2bf(v[lag * K_EIG + k]);
        dst[e] = val;
    }
}

// ---------------------------------------------------------------------------
// Prep 3: transposed bf16 weights. mphiT[o][kd], muT[j][o][i].
// ---------------------------------------------------------------------------
__global__ __launch_bounds__(256)
void k_prep_w(const float* __restrict__ m_phi, const float* __restrict__ m_u,
              unsigned short* __restrict__ mphiT, unsigned short* __restrict__ muT) {
    int id = blockIdx.x * 256 + threadIdx.x;
    if (id < 32768) {                              // 32 o x 1024 kd
        int o = id >> 10, kd = id & 1023;
        mphiT[id] = f2bf(m_phi[kd * D_CH + o]);
    } else if (id < 32768 + 3072) {                // (j*32+o)*32 + i
        int id2 = id - 32768;
        int i = id2 & 31, jo = id2 >> 5;
        int o = jo & 31, j = jo >> 5;
        muT[id2] = f2bf(m_u[(o * D_CH + i) * 3 + j]);
    }
}

// ---------------------------------------------------------------------------
// Kernel 1: spectral conv as tiled Toeplitz GEMM, ping-pong double-buffered.
// grid(b=32, ty=128, kg=2), 256 thr = 8 waves: wave -> (d_half, 4 k values).
// acc[t16 x d16] += A(delta,k)[16x32] * U(b,s-chunk)[32x16], chain over s.
// Stores x_tilde * eig_vals^0.25 as bf16 row-major (b, t, k*32+d).
// ---------------------------------------------------------------------------
#define LOAD_SET(B_, A0_, A1_, A2_, A3_, SC_)                                   \
    do {                                                                        \
        B_ = load_v16(bptr + ((SC_) << 5));                                     \
        const unsigned short* nap_ = abase + (size_t)(ty - 2 * (SC_)) * dstride;\
        A0_ = load_v16(nap_);                                                   \
        A1_ = load_v16(nap_ + 512);                                             \
        A2_ = load_v16(nap_ + 1024);                                            \
        A3_ = load_v16(nap_ + 1536);                                            \
        if ((SC_) + 1 < num_sc) {                                               \
            __builtin_prefetch(bptr + (((SC_) + 1) << 5), 0, 1);                \
            __builtin_prefetch(abase + (size_t)(ty - 2 * ((SC_) + 1)) * dstride,\
                               0, 1);                                           \
        }                                                                       \
    } while (0)

#define DO_WMMA(B_, A0_, A1_, A2_, A3_)                                                                       \
    do {                                                                                                      \
        acc[0] = __builtin_amdgcn_wmma_f32_16x16x32_bf16(false, A0_, false, B_, (short)0, acc[0], false, false);\
        acc[1] = __builtin_amdgcn_wmma_f32_16x16x32_bf16(false, A1_, false, B_, (short)0, acc[1], false, false);\
        acc[2] = __builtin_amdgcn_wmma_f32_16x16x32_bf16(false, A2_, false, B_, (short)0, acc[2], false, false);\
        acc[3] = __builtin_amdgcn_wmma_f32_16x16x32_bf16(false, A3_, false, B_, (short)0, acc[3], false, false);\
    } while (0)

__global__ __launch_bounds__(256)
void k_conv(const unsigned short* __restrict__ Ut,
            const unsigned short* __restrict__ Afrag,
            const float* __restrict__ eig_vals,
            unsigned short* __restrict__ Xs) {
    const int b    = blockIdx.x;
    const int ty   = blockIdx.y;
    const int t0   = ty << 4;
    const int lane = threadIdx.x & 31;
    const int w    = threadIdx.x >> 5;
    const int dh   = w & 1;                        // d half (0/1)
    const int kbase = (blockIdx.z << 4) + ((w >> 1) << 2);
    const int hl   = lane >> 4;
    const int nn   = lane & 15;
    const int d    = (dh << 4) + nn;

    const unsigned short* bptr  = Ut + ((size_t)(b * D_CH + d) << 11) + (hl << 4);
    const unsigned short* abase = Afrag + (((size_t)kbase * 32 + lane) << 4);
    const size_t dstride = (size_t)K_EIG * 32 * 16;   // elements per didx step

    v8f acc[4] = {};

    const int num_sc = (ty >> 1) + 1;              // s-chunks of 32 with s0 <= t0+15

    v16bf bA, a0A, a1A, a2A, a3A;                  // ping
    v16bf bB, a0B, a1B, a2B, a3B;                  // pong

    LOAD_SET(bA, a0A, a1A, a2A, a3A, 0);
    for (int sc = 0; sc < num_sc;) {
        if (sc + 1 < num_sc) LOAD_SET(bB, a0B, a1B, a2B, a3B, sc + 1);
        DO_WMMA(bA, a0A, a1A, a2A, a3A);
        ++sc;
        if (sc >= num_sc) break;
        if (sc + 1 < num_sc) LOAD_SET(bA, a0A, a1A, a2A, a3A, sc + 1);
        DO_WMMA(bB, a0B, a1B, a2B, a3B);
        ++sc;
    }

    float sk[4];
    #pragma unroll
    for (int kk = 0; kk < 4; ++kk) sk[kk] = powf(eig_vals[kbase + kk], 0.25f);

    const int col = (kbase << 5) + d;              // kd = k*32 + d, k step = 32
    #pragma unroll
    for (int r = 0; r < 8; ++r) {                  // C layout: M = r + 8*hl, N = nn
        int t = t0 + r + (hl << 3);
        size_t row = (size_t)(b * L_IN + t) << 10;
        Xs[row + col +  0] = f2bf(acc[0][r] * sk[0]);
        Xs[row + col + 32] = f2bf(acc[1][r] * sk[1]);
        Xs[row + col + 64] = f2bf(acc[2][r] * sk[2]);
        Xs[row + col + 96] = f2bf(acc[3][r] * sk[3]);
    }
}

// ---------------------------------------------------------------------------
// Kernel 2: delta = x_tilde @ m_phi  +  sum_j u[t-j] @ m_u[:,:,j]^T
// grid(b=32, ty=128), 64 thr = 2 waves (one per o-half). 35 WMMAs per wave.
// Writes f32 delta into d_out (consumed in place by the recursion kernel).
// ---------------------------------------------------------------------------
__global__ __launch_bounds__(64)
void k_delta(const unsigned short* __restrict__ Xs,
             const unsigned short* __restrict__ mphiT,
             const unsigned short* __restrict__ Urow,
             const unsigned short* __restrict__ muT,
             float* __restrict__ out) {
    const int b    = blockIdx.x;
    const int ty   = blockIdx.y;
    const int t0   = ty << 4;
    const int lane = threadIdx.x & 31;
    const int oh   = threadIdx.x >> 5;             // o half
    const int hl   = lane >> 4;
    const int nn   = lane & 15;
    const int m    = lane & 15;                    // A-fragment row
    const int o    = (oh << 4) + nn;

    v8f acc = {};

    // phi projection: K contraction over kd=1024 in chunks of 32 (chunk == k)
    const unsigned short* mb = mphiT + ((size_t)o << 10) + (hl << 4);
    const unsigned short* xr = Xs + ((size_t)(b * L_IN + t0 + m) << 10) + (hl << 4);
    for (int c = 0; c < 32; ++c) {
        v16bf a  = load_v16(xr + (c << 5));
        v16bf bb = load_v16(mb + (c << 5));
        acc = __builtin_amdgcn_wmma_f32_16x16x32_bf16(false, a, false, bb, (short)0, acc, false, false);
    }

    // AR term: 3 shifted 16x32x32 GEMMs
    #pragma unroll
    for (int j = 0; j < 3; ++j) {
        int row = t0 + m - j;
        v16bf a;
        if (row >= 0) {
            a = load_v16(Urow + ((size_t)(b * L_IN + row) << 5) + (hl << 4));
        } else {
            #pragma unroll
            for (int e = 0; e < 16; ++e) a[e] = (__bf16)0.0f;
        }
        v16bf bb = load_v16(muT + ((size_t)(j * D_CH + o) << 5) + (hl << 4));
        acc = __builtin_amdgcn_wmma_f32_16x16x32_bf16(false, a, false, bb, (short)0, acc, false, false);
    }

    #pragma unroll
    for (int r = 0; r < 8; ++r) {                  // C layout: M = r + 8*hl, N = nn
        int t = t0 + r + (hl << 3);
        out[((size_t)(b * L_IN + t) << 5) + o] = acc[r];
    }
}

// ---------------------------------------------------------------------------
// Kernel 3: sequential recursion y_t = M1 y_{t-1} + M2 y_{t-2} + delta_t.
// One wave32 per batch, lane = output channel, in-place on d_out.
// Broadcasts via v_readlane (uniform index) -> scalar-operand FMAs, no LDS.
// ---------------------------------------------------------------------------
__global__ __launch_bounds__(32)
void k_scan(const float* __restrict__ m_y, float* __restrict__ out) {
    const int b = blockIdx.x;
    const int o = threadIdx.x;
    float M1[32], M2[32];
    #pragma unroll
    for (int i = 0; i < 32; ++i) {
        M1[i] = m_y[(o * 2 + 0) * D_CH + i];
        M2[i] = m_y[(o * 2 + 1) * D_CH + i];
    }
    float y1 = 0.f, y2 = 0.f;
    float* base = out + (size_t)b * L_IN * D_CH;
    for (int t = 0; t < L_IN; ++t) {
        float dlt = base[t * D_CH + o];
        #pragma unroll
        for (int i = 0; i < 32; ++i) {
            dlt = fmaf(M1[i], bcast_lane(y1, i), dlt);
            dlt = fmaf(M2[i], bcast_lane(y2, i), dlt);
        }
        y2 = y1; y1 = dlt;
        base[t * D_CH + o] = dlt;
    }
}

// ---------------------------------------------------------------------------
extern "C" void kernel_launch(void* const* d_in, const int* in_sizes, int n_in,
                              void* d_out, int out_size, void* d_ws, size_t ws_size,
                              hipStream_t stream) {
    const float* inputs   = (const float*)d_in[0];
    const float* eig_vals = (const float*)d_in[1];
    const float* eig_vecs = (const float*)d_in[2];
    const float* m_u      = (const float*)d_in[3];
    const float* m_phi    = (const float*)d_in[4];
    const float* m_y      = (const float*)d_in[5];
    float* out = (float*)d_out;

    char* ws = (char*)d_ws;
    unsigned short* Ut    = (unsigned short*)(ws);                                     // 4 MB
    unsigned short* Urow  = (unsigned short*)(ws + ((size_t)4  << 20));                // 4 MB
    unsigned short* Afrag = (unsigned short*)(ws + ((size_t)8  << 20));                // 4 MB
    unsigned short* mphiT = (unsigned short*)(ws + ((size_t)12 << 20));                // 64 KB
    unsigned short* muT   = (unsigned short*)(ws + ((size_t)12 << 20) + (128u << 10)); // 6 KB
    unsigned short* Xs    = (unsigned short*)(ws + ((size_t)13 << 20));                // 128 MB

    k_prep_u    <<<dim3((B_BAT * L_IN * D_CH) / 256), 256, 0, stream>>>(inputs, Ut, Urow);
    k_prep_afrag<<<dim3(128, 4), 256, 0, stream>>>(eig_vecs, Afrag);
    k_prep_w    <<<dim3(140), 256, 0, stream>>>(m_phi, m_u, mphiT, muT);
    k_conv      <<<dim3(B_BAT, L_IN / 16, 2), 256, 0, stream>>>(Ut, Afrag, eig_vals, Xs);
    k_delta     <<<dim3(B_BAT, L_IN / 16), 64, 0, stream>>>(Xs, mphiT, Urow, muT, out);
    k_scan      <<<dim3(B_BAT), 32, 0, stream>>>(m_y, out);
}
